// Transform_block_19086834663908
// MI455X (gfx1250) — matrix operand order repaired and checked by reading, not compiled, not run on hardware
//
#include <hip/hip_runtime.h>

// Transformer block (RMSNorm -> MHA+RoPE causal -> residual -> RMSNorm -> SwiGLU -> residual)
// for MI455X / gfx1250: all matmuls via v_wmma_f32_16x16x32_f16 (wave32 WMMA, f32 accum).
// GEMM: 8 waves/block stacked on rows, wave tile 64x64 (16 WMMA accums), shared
// weight tile double-buffered in LDS (global_load_b128 -> ds_store_b128).

static constexpr int Bc  = 2;
static constexpr int Sc  = 2048;
static constexpr int Dc  = 1024;
static constexpr int Hc  = 16;
static constexpr int DKc = 64;
static constexpr int DFFc = 2752;
static constexpr float EPSc = 1e-5f;

typedef __attribute__((ext_vector_type(16))) _Float16 v16h;
typedef __attribute__((ext_vector_type(8)))  float    v8f;

union FragU { v16h h; uint4 q[2]; };

// ---------------------------------------------------------------------------
// WMMA helpers (layouts per CDNA5 ISA §7.12.2)
// ---------------------------------------------------------------------------
__device__ __forceinline__ v8f wmma_f16(v16h a, v16h b, v8f c) {
  // 8 args: (neg_a, A, neg_b, B, c_mod, C, reuse_a, reuse_b)
  return __builtin_amdgcn_wmma_f32_16x16x32_f16(false, a, false, b, (short)0, c,
                                                false, false);
}

// A-matrix fragment, 16x32 f16 (MxK). base points at (row0, k0) of a row-major
// matrix with leading dimension lda. Lane l(=lane&15) holds row l; halves
// 0..7 = K 8*hi+0..7, halves 8..15 = K 8*hi+16..23  -> two 16B loads.
__device__ __forceinline__ v16h load_a_frag(const _Float16* base, int lda) {
  int lane = threadIdx.x & 31;
  int l = lane & 15, hi = lane >> 4;
  const _Float16* p = base + (size_t)l * lda + hi * 8;
  FragU f;
  f.q[0] = *(const uint4*)(p);
  f.q[1] = *(const uint4*)(p + 16);
  return f.h;
}

// B-matrix fragment, 32x16 f16 (KxN) where B[k][n] = W[n][k] and W is
// row-major with leading dimension ldb (i.e. computes X @ W^T).
// base points at W + col0*ldb + k0. Lane l holds column l (= W row col0+l);
// halves 0..7 = K 8*hi+0..7, halves 8..15 = K 8*hi+16..23 -> two 16B loads.
__device__ __forceinline__ v16h load_b_frag(const _Float16* base, int ldb) {
  int lane = threadIdx.x & 31;
  int l = lane & 15, hi = lane >> 4;
  const _Float16* p = base + (size_t)l * ldb + hi * 8;
  FragU f;
  f.q[0] = *(const uint4*)(p);
  f.q[1] = *(const uint4*)(p + 16);
  return f.h;
}

__device__ __forceinline__ void lds_fence() {
  asm volatile("s_wait_dscnt 0" ::: "memory");
}

// ---------------------------------------------------------------------------
// Elementwise / norm kernels
// ---------------------------------------------------------------------------
__global__ __launch_bounds__(256) void k_cvt_f16(const float* __restrict__ src,
                                                 _Float16* __restrict__ dst, int n) {
  int i = blockIdx.x * 256 + threadIdx.x;
  if (i < n) dst[i] = (_Float16)src[i];
}

__global__ __launch_bounds__(256) void k_rmsnorm(const float* __restrict__ X,
                                                 const float* __restrict__ G,
                                                 _Float16* __restrict__ O) {
  int row = blockIdx.x;
  const float* x = X + (size_t)row * Dc;
  float s = 0.f;
  for (int i = threadIdx.x; i < Dc; i += 256) { float v = x[i]; s += v * v; }
  for (int off = 16; off > 0; off >>= 1) s += __shfl_down(s, off, 32);
  __shared__ float red[8];
  if ((threadIdx.x & 31) == 0) red[threadIdx.x >> 5] = s;
  __syncthreads();
  __shared__ float stot;
  if (threadIdx.x == 0) {
    float t = 0.f;
    for (int i = 0; i < 8; i++) t += red[i];
    stot = t;
  }
  __syncthreads();
  float inv = rsqrtf(stot / (float)Dc + EPSc);
  _Float16* o = O + (size_t)row * Dc;
  for (int i = threadIdx.x; i < Dc; i += 256)
    o[i] = (_Float16)(x[i] * inv * G[i]);
}

// RoPE + head-split: Qh [B*S, D] f16 -> Qr [B,H,S,DK] f16
__global__ __launch_bounds__(256) void k_rope(const _Float16* __restrict__ Qh,
                                              const float* __restrict__ cosc,
                                              const float* __restrict__ sinc,
                                              const int* __restrict__ tp,
                                              _Float16* __restrict__ Qr) {
  int idx = blockIdx.x * 256 + threadIdx.x;     // over B*S*H*(DK/2)
  int p = idx & (DKc / 2 - 1);
  int t = idx >> 5;                              // / (DK/2)
  int h = t % Hc; t /= Hc;
  int s = t % Sc; int b = t / Sc;
  if (b >= Bc) return;
  int pos = tp[s]; pos = pos < 0 ? 0 : (pos > Sc - 1 ? Sc - 1 : pos);
  float c = cosc[pos * (DKc / 2) + p];
  float sn = sinc[pos * (DKc / 2) + p];
  size_t src = ((size_t)(b * Sc + s)) * Dc + h * DKc + 2 * p;
  float q0 = (float)Qh[src], q1 = (float)Qh[src + 1];
  size_t dst = (((size_t)(b * Hc + h)) * Sc + s) * DKc + 2 * p;
  Qr[dst]     = (_Float16)(q0 * c - q1 * sn);
  Qr[dst + 1] = (_Float16)(q0 * sn + q1 * c);
}

// V head-split + transpose: Vh [B*S, D] f16 -> Vt [B,H,DK,S] f16
__global__ __launch_bounds__(256) void k_vperm(const _Float16* __restrict__ Vh,
                                               _Float16* __restrict__ Vt) {
  int idx = blockIdx.x * 256 + threadIdx.x;     // over B*H*DK*S
  int s = idx & (Sc - 1);
  int t = idx >> 11;                             // / S
  int dk = t % DKc; t /= DKc;
  int h = t % Hc; int b = t / Hc;
  if (b >= Bc) return;
  Vt[(((size_t)(b * Hc + h)) * DKc + dk) * Sc + s] =
      Vh[((size_t)(b * Sc + s)) * Dc + h * DKc + dk];
}

// SwiGLU: act = silu(gate) * value (f16 in, f16 out)
__global__ __launch_bounds__(256) void k_silu(const _Float16* __restrict__ Gt,
                                              const _Float16* __restrict__ Vl,
                                              _Float16* __restrict__ A, int n) {
  int i = blockIdx.x * 256 + threadIdx.x;
  if (i >= n) return;
  float g = (float)Gt[i];
  float v = (float)Vl[i];
  A[i] = (_Float16)((g / (1.f + __expf(-g))) * v);
}

// ---------------------------------------------------------------------------
// Tiled WMMA GEMM: C[N,M] = A16[N,K] @ W16[M,K]^T  (f32 accumulate)
// Block = 256 thr (8 waves stacked on rows). Block tile 512x64; wave tile
// 64x64 = 4x4 WMMA accumulators (32 flops/byte of fragment traffic).
// The 64x32 weight tile is shared by all 8 waves: staged in LDS (double
// buffered) so the block pulls each weight element from L2 exactly once.
// MODE 0: Cf = acc ; MODE 1: Cf = acc + R ; MODE 2: Ch = (f16)acc
// ---------------------------------------------------------------------------
static constexpr int LDB_PAD = 40;  // halves per LDS row (80B, 16B aligned)

template <int MODE>
__global__ __launch_bounds__(256) void k_gemm(const _Float16* __restrict__ A,
                                              const _Float16* __restrict__ W,
                                              float* __restrict__ Cf,
                                              _Float16* __restrict__ Ch,
                                              const float* __restrict__ R,
                                              int N, int M, int K) {
  __shared__ __align__(16) _Float16 Wlds[2][64][LDB_PAD];

  int w = threadIdx.x >> 5;
  int lane = threadIdx.x & 31;
  int l = lane & 15, hi = lane >> 4;
  int r0 = blockIdx.y * 512 + w * 64;          // wave's output rows
  int c0 = blockIdx.x * 64;                    // block's output cols

  // Per-thread staging slot: 256 threads x 16B == 64x32 halves.
  int srow = threadIdx.x >> 2;                 // 0..63
  int scol = (threadIdx.x & 3) * 8;            // halves 0,8,16,24
  const _Float16* gsrc_base = W + (size_t)(c0 + srow) * K + scol;
  _Float16* ld0 = &Wlds[0][srow][scol];
  _Float16* ld1 = &Wlds[1][srow][scol];

  // preload first weight tile
  uint4 wreg = *(const uint4*)(gsrc_base);
  *(uint4*)ld0 = wreg;
  __syncthreads();

  v8f acc[4][4] = {};
  int buf = 0;
  for (int k0 = 0; k0 < K; k0 += 32) {
    // issue next tile's global load early (lands in regs while WMMAs run)
    uint4 nreg;
    bool more = (k0 + 32) < K;
    if (more) nreg = *(const uint4*)(gsrc_base + k0 + 32);

    v16h a0 = load_a_frag(A + (size_t)(r0     ) * K + k0, K);
    v16h a1 = load_a_frag(A + (size_t)(r0 + 16) * K + k0, K);
    v16h a2 = load_a_frag(A + (size_t)(r0 + 32) * K + k0, K);
    v16h a3 = load_a_frag(A + (size_t)(r0 + 48) * K + k0, K);
    v16h b0 = load_b_frag(&Wlds[buf][ 0][0], LDB_PAD);
    v16h b1 = load_b_frag(&Wlds[buf][16][0], LDB_PAD);
    v16h b2 = load_b_frag(&Wlds[buf][32][0], LDB_PAD);
    v16h b3 = load_b_frag(&Wlds[buf][48][0], LDB_PAD);

    acc[0][0] = wmma_f16(a0, b0, acc[0][0]);
    acc[0][1] = wmma_f16(a0, b1, acc[0][1]);
    acc[0][2] = wmma_f16(a0, b2, acc[0][2]);
    acc[0][3] = wmma_f16(a0, b3, acc[0][3]);
    acc[1][0] = wmma_f16(a1, b0, acc[1][0]);
    acc[1][1] = wmma_f16(a1, b1, acc[1][1]);
    acc[1][2] = wmma_f16(a1, b2, acc[1][2]);
    acc[1][3] = wmma_f16(a1, b3, acc[1][3]);
    acc[2][0] = wmma_f16(a2, b0, acc[2][0]);
    acc[2][1] = wmma_f16(a2, b1, acc[2][1]);
    acc[2][2] = wmma_f16(a2, b2, acc[2][2]);
    acc[2][3] = wmma_f16(a2, b3, acc[2][3]);
    acc[3][0] = wmma_f16(a3, b0, acc[3][0]);
    acc[3][1] = wmma_f16(a3, b1, acc[3][1]);
    acc[3][2] = wmma_f16(a3, b2, acc[3][2]);
    acc[3][3] = wmma_f16(a3, b3, acc[3][3]);

    if (more) {
      *(uint4*)(buf ? ld0 : ld1) = nreg;       // fill the other buffer
    }
    __syncthreads();                           // tile visible to all waves
    buf ^= 1;
  }

#pragma unroll
  for (int i = 0; i < 4; i++)
#pragma unroll
    for (int j = 0; j < 4; j++) {
      int col = c0 + 16 * j + l;
#pragma unroll
      for (int r = 0; r < 8; r++) {
        int row = r0 + 16 * i + 8 * hi + r;    // D-frag: m = r + 8*hi
        size_t idx = (size_t)row * M + col;
        float v = acc[i][j][r];
        if (MODE == 1)      Cf[idx] = v + R[idx];
        else if (MODE == 2) Ch[idx] = (_Float16)v;
        else                Cf[idx] = v;
      }
    }
}

// ---------------------------------------------------------------------------
// Fused causal attention (flash-style, online softmax).
// Grid (B*H, S/32), block 64 (2 waves); each wave owns one 16-query tile.
// Score D-fragment is staged through LDS to (a) run the online softmax and
// (b) relayout into an A-fragment for the P@V WMMAs.
// ---------------------------------------------------------------------------
__global__ __launch_bounds__(64) void k_attn(const _Float16* __restrict__ Qr,
                                             const _Float16* __restrict__ Kr,
                                             const _Float16* __restrict__ Vt,
                                             _Float16* __restrict__ O16) {
  __shared__ float st[2][16][33];                       // score tile (padded)
  __shared__ __align__(16) _Float16 pt[2][16][32];      // P tile (A-frag layout)
  __shared__ float al[2][16];                           // per-row alpha / inv-sum

  int bh = blockIdx.x;
  int b = bh / Hc, h = bh % Hc;
  int w = threadIdx.x >> 5;
  int qt = blockIdx.y * 2 + w;
  int lane = threadIdx.x & 31;
  int l = lane & 15, hi = lane >> 4;
  int q0 = qt * 16;
  const float scale = 0.125f;                           // 1/sqrt(DK=64)

  const _Float16* Qhd = Qr + ((size_t)bh * Sc + q0) * DKc;
  v16h aq0 = load_a_frag(Qhd, DKc);                     // K-chunk 0..31
  v16h aq1 = load_a_frag(Qhd + 32, DKc);                // K-chunk 32..63

  v8f o[4] = {};
  float mrow = -1e30f, lrow = 0.f;                      // row l stats (dup x2)

  int nk32 = (q0 + 16 + 31) / 32;                       // causal 32-key tiles
  for (int kt = 0; kt < nk32; kt++) {
    int key0 = kt * 32;
    const _Float16* Khd = Kr + ((size_t)bh * Sc + key0) * DKc;
#pragma unroll
    for (int j = 0; j < 2; j++) {
      v8f sc = {};
      v16h bk0 = load_b_frag(Khd + (size_t)(16 * j) * DKc, DKc);
      v16h bk1 = load_b_frag(Khd + (size_t)(16 * j) * DKc + 32, DKc);
      sc = wmma_f16(aq0, bk0, sc);
      sc = wmma_f16(aq1, bk1, sc);
      int keyg = key0 + 16 * j + l;
#pragma unroll
      for (int r = 0; r < 8; r++) {
        int m = r + 8 * hi;
        float v = sc[r] * scale;
        if (keyg > q0 + m) v = -1e30f;                  // causal mask
        st[w][m][16 * j + l] = v;
      }
    }
    lds_fence();
    // Online softmax: row m = l, done redundantly by lanes l and l+16.
    {
      int m = l;
      float tmax = -1e30f;
      for (int kk = 0; kk < 32; kk++) tmax = fmaxf(tmax, st[w][m][kk]);
      float mnew = fmaxf(mrow, tmax);
      float alpha = __expf(mrow - mnew);
      float ls = 0.f;
      for (int kk = 0; kk < 32; kk++) {
        float p = __expf(st[w][m][kk] - mnew);
        ls += p;
        pt[w][m][kk] = (_Float16)p;
      }
      lrow = lrow * alpha + ls;
      mrow = mnew;
      al[w][m] = alpha;
    }
    lds_fence();
    v16h pa = load_a_frag(&pt[w][0][0], 32);
    float av[8];
#pragma unroll
    for (int r = 0; r < 8; r++) av[r] = al[w][r + 8 * hi];
    const _Float16* Vhd = Vt + (size_t)bh * DKc * Sc + key0;
#pragma unroll
    for (int t = 0; t < 4; t++) {
      v16h bv = load_b_frag(Vhd + (size_t)(16 * t) * Sc, Sc);
#pragma unroll
      for (int r = 0; r < 8; r++) o[t][r] *= av[r];
      o[t] = wmma_f16(pa, bv, o[t]);
    }
  }

  al[w][l] = 1.f / lrow;                                // reuse for inv-sum
  lds_fence();
  size_t rowb = (size_t)b * Sc + q0;
#pragma unroll
  for (int t = 0; t < 4; t++)
#pragma unroll
    for (int r = 0; r < 8; r++) {
      int m = r + 8 * hi;
      O16[(rowb + m) * Dc + h * DKc + 16 * t + l] = (_Float16)(o[t][r] * al[w][m]);
    }
}

// ---------------------------------------------------------------------------
// Host-side orchestration
// ---------------------------------------------------------------------------
extern "C" void kernel_launch(void* const* d_in, const int* in_sizes, int n_in,
                              void* d_out, int out_size, void* d_ws, size_t ws_size,
                              hipStream_t stream) {
  (void)in_sizes; (void)n_in; (void)out_size; (void)ws_size;
  const float* x    = (const float*)d_in[0];
  const float* wq   = (const float*)d_in[1];
  const float* wk   = (const float*)d_in[2];
  const float* wv   = (const float*)d_in[3];
  const float* wo   = (const float*)d_in[4];
  const float* g1   = (const float*)d_in[5];
  const float* g2   = (const float*)d_in[6];
  const float* w1   = (const float*)d_in[7];
  const float* w2   = (const float*)d_in[8];
  const float* w3   = (const float*)d_in[9];
  const float* cosc = (const float*)d_in[10];
  const float* sinc = (const float*)d_in[11];
  const int*   tpos = (const int*)d_in[12];
  float* out = (float*)d_out;

  const int NTOK = Bc * Sc;                 // 4096 rows
  char* ws = (char*)d_ws;
  size_t off = 0;
  auto alloc = [&](size_t bytes) -> char* {
    char* p = ws + off;
    off += (bytes + 255) & ~(size_t)255;
    return p;
  };
  _Float16* wq16 = (_Float16*)alloc((size_t)Dc * Dc * 2);
  _Float16* wk16 = (_Float16*)alloc((size_t)Dc * Dc * 2);
  _Float16* wv16 = (_Float16*)alloc((size_t)Dc * Dc * 2);
  _Float16* wo16 = (_Float16*)alloc((size_t)Dc * Dc * 2);
  _Float16* w116 = (_Float16*)alloc((size_t)DFFc * Dc * 2);
  _Float16* w316 = (_Float16*)alloc((size_t)DFFc * Dc * 2);
  _Float16* w216 = (_Float16*)alloc((size_t)Dc * DFFc * 2);
  _Float16* h16  = (_Float16*)alloc((size_t)NTOK * Dc * 2);
  _Float16* Qh   = (_Float16*)alloc((size_t)NTOK * Dc * 2);
  _Float16* Kh   = (_Float16*)alloc((size_t)NTOK * Dc * 2);
  _Float16* Vh   = (_Float16*)alloc((size_t)NTOK * Dc * 2);
  _Float16* Qr   = (_Float16*)alloc((size_t)NTOK * Dc * 2);
  _Float16* Kr   = (_Float16*)alloc((size_t)NTOK * Dc * 2);
  _Float16* Vt   = (_Float16*)alloc((size_t)NTOK * Dc * 2);
  _Float16* O16  = (_Float16*)alloc((size_t)NTOK * Dc * 2);
  float*    out1 = (float*)   alloc((size_t)NTOK * Dc * 4);
  _Float16* h216 = (_Float16*)alloc((size_t)NTOK * Dc * 2);
  _Float16* gt16 = (_Float16*)alloc((size_t)NTOK * DFFc * 2);
  _Float16* vl16 = (_Float16*)alloc((size_t)NTOK * DFFc * 2);
  _Float16* ac16 = (_Float16*)alloc((size_t)NTOK * DFFc * 2);

  auto cvt = [&](const float* s, _Float16* d, int n) {
    k_cvt_f16<<<(n + 255) / 256, 256, 0, stream>>>(s, d, n);
  };
  cvt(wq, wq16, Dc * Dc);
  cvt(wk, wk16, Dc * Dc);
  cvt(wv, wv16, Dc * Dc);
  cvt(wo, wo16, Dc * Dc);
  cvt(w1, w116, DFFc * Dc);
  cvt(w3, w316, DFFc * Dc);
  cvt(w2, w216, Dc * DFFc);

  // h = rmsnorm(x, g1)
  k_rmsnorm<<<NTOK, 256, 0, stream>>>(x, g1, h16);

  dim3 gQKV(Dc / 64, NTOK / 512);
  k_gemm<2><<<gQKV, 256, 0, stream>>>(h16, wq16, nullptr, Qh, nullptr, NTOK, Dc, Dc);
  k_gemm<2><<<gQKV, 256, 0, stream>>>(h16, wk16, nullptr, Kh, nullptr, NTOK, Dc, Dc);
  k_gemm<2><<<gQKV, 256, 0, stream>>>(h16, wv16, nullptr, Vh, nullptr, NTOK, Dc, Dc);

  int nrope = Bc * Sc * Hc * (DKc / 2);
  k_rope<<<(nrope + 255) / 256, 256, 0, stream>>>(Qh, cosc, sinc, tpos, Qr);
  k_rope<<<(nrope + 255) / 256, 256, 0, stream>>>(Kh, cosc, sinc, tpos, Kr);
  int nvp = Bc * Hc * DKc * Sc;
  k_vperm<<<(nvp + 255) / 256, 256, 0, stream>>>(Vh, Vt);

  dim3 gAtt(Bc * Hc, Sc / 32);
  k_attn<<<gAtt, 64, 0, stream>>>(Qr, Kr, Vt, O16);

  // out1 = O @ wo^T + x
  k_gemm<1><<<gQKV, 256, 0, stream>>>(O16, wo16, out1, nullptr, x, NTOK, Dc, Dc);

  // h2 = rmsnorm(out1, g2)
  k_rmsnorm<<<NTOK, 256, 0, stream>>>(out1, g2, h216);

  dim3 gFF(DFFc / 64, NTOK / 512);
  k_gemm<2><<<gFF, 256, 0, stream>>>(h216, w116, nullptr, gt16, nullptr, NTOK, DFFc, Dc);
  k_gemm<2><<<gFF, 256, 0, stream>>>(h216, w316, nullptr, vl16, nullptr, NTOK, DFFc, Dc);

  int nsl = NTOK * DFFc;
  k_silu<<<(nsl + 255) / 256, 256, 0, stream>>>(gt16, vl16, ac16, nsl);

  // out = act @ w2^T + out1
  k_gemm<1><<<gQKV, 256, 0, stream>>>(ac16, w216, out, nullptr, out1, NTOK, Dc, DFFc);
}
